// AdaptiveAngleConv_74028056314165
// MI455X (gfx1250) — compile-verified, matrix operand order checked by reading, not run on hardware
//
#include <hip/hip_runtime.h>
#include <hip/hip_bf16.h>

// ---------------------------------------------------------------------------
// AdaptiveAngleConv on MI455X (gfx1250): implicit-GEMM 3x3 conv, 4 rotated
// kernel variants sharing one weight copy in LDS. f16 WMMA, f32 accumulate.
//   out[a][b][co][h][w] = bias[co] +
//       sum_{ci,t} x[b][ci][h+r(t)-1][w+s(t)-1] * W[co][ci][PERMS[a][t]]
// GEMM view per workgroup: M=64 (co), N=64 (pixels of one row), K=576=(t,ci).
// ---------------------------------------------------------------------------

typedef __attribute__((ext_vector_type(16))) _Float16 v16h;
typedef __attribute__((ext_vector_type(8)))  _Float16 v8h;
typedef __attribute__((ext_vector_type(8)))  float    v8f;

#define CIN     64
#define COUT    64
#define HDIM    128
#define WDIM    128
#define BATCH   16
#define NANG    4
#define KTOT    576          // 9 taps * 64 ci
#define WPITCH  584          // halfs per weight row: 16B aligned, conflict-free
#define XPITCH  72           // halfs per (r,col) row:  16B aligned, low-conflict
#define XCOLS   66           // 64 pixels + 2 halo
#define W_LDS_HALFS (COUT * WPITCH)          // 37376
#define X_LDS_HALFS (3 * XCOLS * XPITCH)     // 14256
#define SMEM_BYTES  ((W_LDS_HALFS + X_LDS_HALFS) * 2)   // 103,264 B < 320 KB

__constant__ int PERMS_DEV[NANG][9] = {
    {0, 1, 2, 3, 4, 5, 6, 7, 8},   //   0 deg
    {3, 0, 1, 6, 4, 2, 7, 8, 5},   //  45 deg
    {6, 3, 0, 7, 4, 1, 8, 5, 2},   //  90 deg
    {7, 6, 3, 8, 4, 0, 5, 2, 1},   // 135 deg
};

__device__ __forceinline__ v16h cat8(v8h lo, v8h hi) {
    return __builtin_shufflevector(lo, hi, 0, 1, 2, 3, 4, 5, 6, 7,
                                           8, 9, 10, 11, 12, 13, 14, 15);
}

__global__ void __launch_bounds__(256, 1)
AdaptiveAngleConv_74028056314165_kernel(const float* __restrict__ x,
                                        const float* __restrict__ weight,
                                        const float* __restrict__ bias,
                                        float* __restrict__ out) {
    extern __shared__ char smem[];
    _Float16* Wlds = reinterpret_cast<_Float16*>(smem);
    _Float16* Xlds = Wlds + W_LDS_HALFS;

    const int tid  = threadIdx.x;
    const int lane = tid & 31;
    const int wave = tid >> 5;
    const int ang  = wave & 3;      // angle handled by this wave
    const int nhlf = wave >> 2;     // which 32-pixel half of the 64-pixel tile
    const int lm   = lane & 15;     // row (A) or column (B/C) within tile
    const int sel  = lane >> 4;     // which K-half group this lane holds

    const int w0 = blockIdx.x * 64; // pixel tile start
    const int h  = blockIdx.y;
    const int b  = blockIdx.z;

    // ---- Stage rotated-invariant weights: Wlds[co][t*64+ci] (f16) ----------
    // Global weight is [co][ci][t] row-major -> consecutive i is coalesced.
    for (int i = tid; i < COUT * CIN * 9; i += 256) {
        const int co  = i / (CIN * 9);
        const int rem = i - co * (CIN * 9);
        const int ci  = rem / 9;
        const int t   = rem - ci * 9;
        Wlds[co * WPITCH + t * CIN + ci] = (_Float16)weight[i];
    }

    // ---- Stage x halo tile: Xlds[(r*66+col)*72 + ci] (f16, zero-padded) ----
    // i ordered (ci, r, col) so consecutive lanes read consecutive cols.
    for (int i = tid; i < CIN * 3 * XCOLS; i += 256) {
        const int ci  = i / (3 * XCOLS);
        const int rem = i - ci * (3 * XCOLS);
        const int r   = rem / XCOLS;
        const int col = rem - r * XCOLS;
        const int hh  = h + r - 1;
        const int ww  = w0 + col - 1;
        float v = 0.0f;
        if (hh >= 0 && hh < HDIM && ww >= 0 && ww < WDIM)
            v = x[(((size_t)b * CIN + ci) * HDIM + hh) * WDIM + ww];
        Xlds[(r * XCOLS + col) * XPITCH + ci] = (_Float16)v;
    }

    __syncthreads();

    // Hoist this wave's tap permutation out of the K loop.
    int permA[9];
#pragma unroll
    for (int t = 0; t < 9; ++t) permA[t] = PERMS_DEV[ang][t];

    v8f acc[4][2];
#pragma unroll
    for (int mt = 0; mt < 4; ++mt)
#pragma unroll
        for (int nt = 0; nt < 2; ++nt)
            acc[mt][nt] = v8f{0.f, 0.f, 0.f, 0.f, 0.f, 0.f, 0.f, 0.f};

    // ---- K loop: 18 chunks of 32 (tap-major, ci-minor) ---------------------
#pragma unroll
    for (int kk = 0; kk < 18; ++kk) {
        const int rs  = kk >> 1;          // tap index on the x side
        const int ci0 = (kk & 1) << 5;    // 0 or 32
        const int r   = rs / 3;
        const int s   = rs - r * 3;

        // B fragments (x patches), shared across the 4 angles.
        v16h bfrag[2];
#pragma unroll
        for (int nt = 0; nt < 2; ++nt) {
            const int n   = nhlf * 32 + nt * 16 + lm;  // pixel within tile
            const int col = n + s;                     // 0..65 in halo coords
            const _Float16* p =
                &Xlds[(r * XCOLS + col) * XPITCH + ci0 + sel * 8];
            const v8h lo = *reinterpret_cast<const v8h*>(p);
            const v8h hi = *reinterpret_cast<const v8h*>(p + 16);
            bfrag[nt] = cat8(lo, hi);
        }

        // A fragments: permuted tap row for this wave's angle.
        const int kbA = permA[rs] * CIN + ci0;
#pragma unroll
        for (int mt = 0; mt < 4; ++mt) {
            const int m = mt * 16 + lm;
            const _Float16* q = &Wlds[m * WPITCH + kbA + sel * 8];
            const v8h lo = *reinterpret_cast<const v8h*>(q);
            const v8h hi = *reinterpret_cast<const v8h*>(q + 16);
            const v16h afrag = cat8(lo, hi);
            acc[mt][0] = __builtin_amdgcn_wmma_f32_16x16x32_f16(
                false, afrag, false, bfrag[0], (short)0, acc[mt][0],
                false, false);
            acc[mt][1] = __builtin_amdgcn_wmma_f32_16x16x32_f16(
                false, afrag, false, bfrag[1], (short)0, acc[mt][1],
                false, false);
        }
    }

    // ---- Epilogue: bias + coalesced f32 stores -----------------------------
    // C layout: VGPR v -> row (v + 8*sel), lane%16 -> column.
#pragma unroll
    for (int mt = 0; mt < 4; ++mt) {
#pragma unroll
        for (int nt = 0; nt < 2; ++nt) {
            const int n = (nhlf * 2 + nt) * 16 + lm;
            const int w = w0 + n;
#pragma unroll
            for (int v = 0; v < 8; ++v) {
                const int co = mt * 16 + sel * 8 + v;
                const size_t o =
                    ((((size_t)ang * BATCH + b) * COUT + co) * HDIM + h) * WDIM
                    + w;
                out[o] = acc[mt][nt][v] + bias[co];
            }
        }
    }
}

extern "C" void kernel_launch(void* const* d_in, const int* in_sizes, int n_in,
                              void* d_out, int out_size, void* d_ws,
                              size_t ws_size, hipStream_t stream) {
    const float* x      = (const float*)d_in[0];
    const float* weight = (const float*)d_in[1];
    const float* bias   = (const float*)d_in[2];
    float* out          = (float*)d_out;

    (void)in_sizes; (void)n_in; (void)out_size; (void)d_ws; (void)ws_size;

    // 103 KB dynamic LDS (> default 64 KB cap) — raise the limit every call
    // (idempotent, host-side, graph-capture safe).
    hipFuncSetAttribute(
        reinterpret_cast<const void*>(&AdaptiveAngleConv_74028056314165_kernel),
        hipFuncAttributeMaxDynamicSharedMemorySize, SMEM_BYTES);

    dim3 grid(WDIM / 64, HDIM, BATCH);   // (2, 128, 16)
    dim3 block(256);                      // 8 waves: 4 angles x 2 pixel halves
    AdaptiveAngleConv_74028056314165_kernel<<<grid, block, SMEM_BYTES, stream>>>(
        x, weight, bias, out);
}
